// ResNet18_delta_SCNN_51694226375416
// MI455X (gfx1250) — compile-verified, matrix operand order checked by reading, not run on hardware
//
#include <hip/hip_runtime.h>

// ---------------------------------------------------------------------------
// CDNA5 (gfx1250) wave32 WMMA types
// ---------------------------------------------------------------------------
typedef __attribute__((ext_vector_type(16))) __bf16 v16bf;
typedef __attribute__((ext_vector_type(8)))  float  v8f;

// fp32 -> bf16 round-to-nearest-even
__device__ __forceinline__ __bf16 f2bf(float f) {
  union { float f; unsigned u; } c; c.f = f;
  unsigned r = c.u + 0x7FFFu + ((c.u >> 16) & 1u);
  unsigned short h = (unsigned short)(r >> 16);
  return __builtin_bit_cast(__bf16, h);
}

__device__ __forceinline__ __bf16 bf_zero() {
  return __builtin_bit_cast(__bf16, (unsigned short)0);
}

__device__ __forceinline__ v8f wmma_bf16(v16bf A, v16bf B, v8f C) {
  return __builtin_amdgcn_wmma_f32_16x16x32_bf16(false, A, false, B, (short)0, C,
                                                 false, false);
}

// 16x32 bf16 A/B fragment K index for (element e in 0..15, lane-half)
// per ISA 7.12.2: VGPR v holds K pair; v0..3 -> K=2v(+8*half), v4..7 -> +16
__device__ __host__ __forceinline__ int frag_k(int e, int half) {
  int v = e >> 1;
  int base = (v < 4) ? (2 * v) : (2 * v + 8);
  return base + 8 * half + (e & 1);
}

// ---------------------------------------------------------------------------
// Streaming fp32 -> bf16 conversion (one pass per conv input activation).
// ---------------------------------------------------------------------------
__global__ void cvt_bf16_kernel(const float* __restrict__ x, __bf16* __restrict__ y,
                                int nquads) {
  int i = blockIdx.x * blockDim.x + threadIdx.x;
  if (i >= nquads) return;
  float4 v = ((const float4*)x)[i];
  __bf16 o[4] = {f2bf(v.x), f2bf(v.y), f2bf(v.z), f2bf(v.w)};
  ((ulong1*)y)[i] = *(ulong1*)o;     // 8-byte store of 4 bf16
}

// ---------------------------------------------------------------------------
// Weight pre-pack: fp32 [Cout][K] (OIHW flat) -> bf16 fragment-ordered
// layout [ntile][kc][lane][16], K zero-padded to multiple of 32.
// ---------------------------------------------------------------------------
__global__ void pack_w_kernel(const float* __restrict__ w, __bf16* __restrict__ wp,
                              int Cout, int K, int KC, int total) {
  int i = blockIdx.x * blockDim.x + threadIdx.x;
  if (i >= total) return;
  int e    = i & 15;
  int lane = (i >> 4) & 31;
  int rest = i >> 9;                 // ntile*KC + kc
  int kc    = rest % KC;
  int ntile = rest / KC;
  int half  = lane >> 4;
  int n = ntile * 16 + (lane & 15);  // Cout is always a multiple of 16 here
  int k = kc * 32 + frag_k(e, half);
  float val = (k < K) ? w[(size_t)n * K + k] : 0.0f;
  wp[i] = f2bf(val);
}

// ---------------------------------------------------------------------------
// Implicit-GEMM conv2d with v_wmma_f32_16x16x32_bf16.
//   M = B*Hout*Wout, tiled 64 pixels/block (4 sub-tiles of 16).
//   N = Cout, 16 channels per wave (up to 8 waves/block).
//   K = Cin*KH*KW in chunks of 32 (bf16 A gathered from pre-converted bf16
//   activations; fragment-packed bf16 B), fp32 accumulate.
// A fragments double-buffered in LDS, built cooperatively by waves,
// ONE barrier per K-chunk. KH/KW/STRIDE template constants -> all hot-loop
// div/mod are by compile-time constants.
// Epilogue: +bias, *bn_s+bn_b, +residual(fp32), relu (flag controlled).
// ---------------------------------------------------------------------------
#define CF_BIAS 1
#define CF_BN   2
#define CF_RELU 4
#define CF_RES  8

struct ConvArgs {
  const __bf16* x;        // bf16 input activation [B,Cin,H,W]
  const __bf16* wp;       // fragment-packed weights
  const float*  bias;
  const float*  bns;
  const float*  bnb;
  const float*  res;      // fp32 residual [B,Cout,Hout,Wout]
  float*        y;
  int B, Cin, H, W, Cout, Hout, Wout, pad, KC, flags;
};

template <int KH, int KW, int STRIDE>
__global__ __launch_bounds__(256) void conv2d_wmma_kernel(ConvArgs a) {
  constexpr int KHW = KH * KW;
  const int lane  = threadIdx.x & 31;
  const int wave  = threadIdx.x >> 5;
  const int nwav  = blockDim.x >> 5;
  const int ntile = blockIdx.y * nwav + wave;
  const int half  = lane >> 4;
  const int mrow  = lane & 15;
  const int Ktot  = a.Cin * KHW;
  const int HWo   = a.Hout * a.Wout;
  const int Mtot  = a.B * HWo;
  const int H = a.H, W = a.W;

  __shared__ __align__(32) __bf16 As[2][4][512];   // [buf][sub][lane*16+e]

  // Per-subtile A-row decomposition for this lane (hoisted out of K loop)
  bool s_v[4];
  int  s_ih0[4], s_iw0[4];
  const __bf16* s_xb[4];
  #pragma unroll
  for (int s = 0; s < 4; ++s) {
    int m = (blockIdx.x * 4 + s) * 16 + mrow;
    s_v[s] = m < Mtot;
    int mm = s_v[s] ? m : 0;
    int b = mm / HWo; int r = mm % HWo;
    s_ih0[s] = (r / a.Wout) * STRIDE - a.pad;
    s_iw0[s] = (r % a.Wout) * STRIDE - a.pad;
    s_xb[s]  = a.x + (size_t)b * a.Cin * H * W;
  }

  auto build = [&](int buf, int kc) {
    for (int s = wave; s < 4; s += nwav) {
      const __bf16* xb = s_xb[s];
      const int ih0 = s_ih0[s], iw0 = s_iw0[s];
      #pragma unroll
      for (int e = 0; e < 16; ++e) {
        int k = kc * 32 + frag_k(e, half);
        __bf16 val = bf_zero();
        if (s_v[s] && k < Ktot) {
          int ci = k / KHW;           // compile-time divisor
          int r  = k - ci * KHW;
          int kh = r / KW;            // compile-time divisor
          int kw = r - kh * KW;
          int ih = ih0 + kh, iw = iw0 + kw;
          if (ih >= 0 && ih < H && iw >= 0 && iw < W)
            val = xb[((size_t)ci * H + ih) * W + iw];
        }
        As[buf][s][lane * 16 + e] = val;
      }
    }
  };

  v8f acc[4] = {v8f{}, v8f{}, v8f{}, v8f{}};
  build(0, 0);
  for (int kc = 0; kc < a.KC; ++kc) {
    __syncthreads();                       // publishes build(kc), retires reads(kc-1)
    if (kc + 1 < a.KC) {
      build((kc + 1) & 1, kc + 1);         // overlap next A-stage with this WMMA
      __builtin_prefetch(&a.wp[((size_t)(ntile * a.KC + kc + 1) * 32 + lane) * 16],
                         0, 0);
    }
    v16bf bf = *(const v16bf*)&a.wp[((size_t)(ntile * a.KC + kc) * 32 + lane) * 16];
    const __bf16* ap = &As[kc & 1][0][lane * 16];
    #pragma unroll
    for (int s = 0; s < 4; ++s)
      acc[s] = wmma_bf16(*(const v16bf*)(ap + s * 512), bf, acc[s]);
  }

  // Epilogue. D layout: lanes 0-15 -> M=j, lanes 16-31 -> M=j+8; N = lane&15.
  int n = ntile * 16 + (lane & 15);
  float bv  = (a.flags & CF_BIAS) ? a.bias[n] : 0.0f;
  float sv  = (a.flags & CF_BN)   ? a.bns[n]  : 1.0f;
  float sbv = (a.flags & CF_BN)   ? a.bnb[n]  : 0.0f;
  #pragma unroll
  for (int s = 0; s < 4; ++s) {
    #pragma unroll
    for (int j = 0; j < 8; ++j) {
      int m = (blockIdx.x * 4 + s) * 16 + j + 8 * half;
      if (m >= Mtot) continue;
      int b = m / HWo; int r = m % HWo;
      int oh = r / a.Wout, ow = r % a.Wout;
      float v = (acc[s][j] + bv) * sv + sbv;
      size_t oi = (((size_t)b * a.Cout + n) * a.Hout + oh) * a.Wout + ow;
      if (a.flags & CF_RES) v += a.res[oi];
      if (a.flags & CF_RELU) v = fmaxf(v, 0.0f);
      a.y[oi] = v;
    }
  }
}

// ---------------------------------------------------------------------------
// 3x3 stride-2 pad-1 max pool
// ---------------------------------------------------------------------------
__global__ void maxpool_kernel(const float* __restrict__ x, float* __restrict__ y,
                               int BC, int H, int W, int Ho, int Wo) {
  int i = blockIdx.x * blockDim.x + threadIdx.x;
  int total = BC * Ho * Wo;
  if (i >= total) return;
  int ow = i % Wo; int r = i / Wo;
  int oh = r % Ho; int bc = r / Ho;
  float m = -3.4e38f;
  #pragma unroll
  for (int kh = 0; kh < 3; ++kh) {
    int ih = oh * 2 - 1 + kh;
    if (ih < 0 || ih >= H) continue;
    #pragma unroll
    for (int kw = 0; kw < 3; ++kw) {
      int iw = ow * 2 - 1 + kw;
      if (iw < 0 || iw >= W) continue;
      m = fmaxf(m, x[((size_t)bc * H + ih) * W + iw]);
    }
  }
  y[i] = m;
}

// ---------------------------------------------------------------------------
// ConvTranspose2d(k=3, stride=2, pad=1, output_pad=1) == conv over 2x-dilated
// input with padding (1,2). Parity-dependent taps -> direct kernel.
// ---------------------------------------------------------------------------
__global__ void convt_kernel(const float* __restrict__ x, const float* __restrict__ w,
                             const float* __restrict__ bias, float* __restrict__ y,
                             int B, int Cin, int H, int W, int Cout, int Ho, int Wo,
                             int relu) {
  int i = blockIdx.x * blockDim.x + threadIdx.x;
  int total = B * Cout * Ho * Wo;
  if (i >= total) return;
  int ow = i % Wo; int r = i / Wo;
  int oh = r % Ho; r /= Ho;
  int co = r % Cout; int b = r / Cout;
  float acc = bias[co];
  #pragma unroll
  for (int kh = 0; kh < 3; ++kh) {
    int d = oh + kh - 1;
    if (d < 0 || (d & 1)) continue;
    int ih = d >> 1; if (ih >= H) continue;
    #pragma unroll
    for (int kw = 0; kw < 3; ++kw) {
      int e = ow + kw - 1;
      if (e < 0 || (e & 1)) continue;
      int iw = e >> 1; if (iw >= W) continue;
      const float* xp = &x[(((size_t)b * Cin) * H + ih) * W + iw];
      const float* wp = &w[((size_t)co * Cin * 3 + kh) * 3 + kw];
      for (int ci = 0; ci < Cin; ++ci)
        acc += xp[(size_t)ci * H * W] * wp[(size_t)ci * 9];
    }
  }
  if (relu) acc = fmaxf(acc, 0.0f);
  y[i] = acc;
}

// ---------------------------------------------------------------------------
// One SCNN message-passing step:  F[dst] += relu(conv1d9(F[src], w))
// vert=1: slices along H, conv over W (M = B*W = 320, 20 M-tiles)
// vert=0: slices along W, conv over H (M = B*H = 184, 12 M-tiles)
// N = 128 (8 waves = 8 N-tiles), K = 128*9 = 1152 (KC = 36).
// Double-buffered A staging, one barrier per K-chunk. In-place on fp32.
// ---------------------------------------------------------------------------
__global__ __launch_bounds__(256) void scnn_step_kernel(float* __restrict__ Fm,
                                                        const __bf16* __restrict__ wp,
                                                        int vert, int src, int dst) {
  const int C = 128, H = 92, W = 160, KC = 36;
  const int lane = threadIdx.x & 31;
  const int wave = threadIdx.x >> 5;         // == ntile
  const int half = lane >> 4;
  const int mrow = lane & 15;
  const int L = vert ? W : H;
  const int M = 2 * L;
  const int mtile = blockIdx.x;

  __shared__ __align__(32) __bf16 As[2][512];

  int am = mtile * 16 + mrow;
  bool amv = am < M;
  int ab  = amv ? am / L : 0;
  int apo = amv ? am % L : 0;

  auto build = [&](int buf, int kc) {
    #pragma unroll
    for (int e = 0; e < 16; ++e) {
      int k = kc * 32 + frag_k(e, half);
      int ci = k / 9, t = k % 9;          // compile-time divisors
      int p = apo + t - 4;
      float val = 0.0f;
      if (amv && p >= 0 && p < L) {
        val = vert ? Fm[(((size_t)ab * C + ci) * H + src) * W + p]
                   : Fm[(((size_t)ab * C + ci) * H + p) * W + src];
      }
      As[buf][lane * 16 + e] = f2bf(val);
    }
  };

  v8f acc = {};
  if (wave == 0) build(0, 0);
  for (int kc = 0; kc < KC; ++kc) {
    __syncthreads();
    if (wave == 0 && kc + 1 < KC) build((kc + 1) & 1, kc + 1);
    v16bf af = *(const v16bf*)&As[kc & 1][lane * 16];
    v16bf bf = *(const v16bf*)&wp[((size_t)(wave * KC + kc) * 32 + lane) * 16];
    acc = wmma_bf16(af, bf, acc);
  }

  int n = wave * 16 + (lane & 15);
  #pragma unroll
  for (int j = 0; j < 8; ++j) {
    int m = mtile * 16 + j + 8 * half;
    if (m >= M) continue;
    int b = m / L, pos = m % L;
    float rlu = fmaxf(acc[j], 0.0f);
    size_t oi = vert ? (((size_t)b * C + n) * H + dst) * W + pos
                     : (((size_t)b * C + n) * H + pos) * W + dst;
    Fm[oi] += rlu;
  }
}

// ---------------------------------------------------------------------------
// Host orchestration
// ---------------------------------------------------------------------------
extern "C" void kernel_launch(void* const* d_in, const int* in_sizes, int n_in,
                              void* d_out, int out_size, void* d_ws, size_t ws_size,
                              hipStream_t stream) {
  (void)in_sizes; (void)n_in; (void)out_size; (void)ws_size;
  const int B = 2;
  auto F = [&](int i) { return (const float*)d_in[i]; };

  // ---- workspace: 4 fp32 ping-pong buffers + bf16 activation staging +
  //      fragment-packed bf16 weights
  const size_t PBUF = (size_t)2 * 512 * 92 * 160;   // 15,073,280 elems (== 2*128*184*320)
  float* P0 = (float*)d_ws;
  float* P1 = P0 + PBUF;
  float* P2 = P1 + PBUF;
  float* P3 = P2 + PBUF;
  __bf16* XB    = (__bf16*)(P3 + PBUF);             // bf16 copy of current conv input
  __bf16* wbase = XB + PBUF;
  size_t woff = 0;

  auto pack = [&](const float* w, int Cout, int K) -> const __bf16* {
    int KC = (K + 31) / 32;
    size_t total = (size_t)(Cout / 16) * KC * 512;
    __bf16* dst = wbase + woff;
    woff += total;
    pack_w_kernel<<<dim3((unsigned)((total + 255) / 256)), 256, 0, stream>>>(
        w, dst, Cout, K, KC, (int)total);
    return dst;
  };

  auto conv = [&](const float* x, const __bf16* wp, const float* bias,
                  const float* s, const float* sb, const float* res, float* y,
                  int Cin, int H, int W, int Cout, int Ho, int Wo,
                  int KH, int st, int pd, int flags) {
    // stage input activation as bf16 (one streaming pass)
    int nquads = (B * Cin * H * W) / 4;
    cvt_bf16_kernel<<<(nquads + 255) / 256, 256, 0, stream>>>(x, XB, nquads);

    ConvArgs a;
    a.x = XB; a.wp = wp; a.bias = bias; a.bns = s; a.bnb = sb; a.res = res; a.y = y;
    a.B = B; a.Cin = Cin; a.H = H; a.W = W; a.Cout = Cout; a.Hout = Ho; a.Wout = Wo;
    a.pad = pd; a.KC = (Cin * KH * KH + 31) / 32; a.flags = flags;
    int Mtiles = (B * Ho * Wo + 63) / 64;          // 64 pixels per block
    int Nt = Cout / 16;
    int nw = Nt < 8 ? Nt : 8;
    dim3 grid((unsigned)Mtiles, (unsigned)(Nt / nw));
    if (KH == 7)
      conv2d_wmma_kernel<7, 7, 2><<<grid, nw * 32, 0, stream>>>(a);
    else if (KH == 1)
      conv2d_wmma_kernel<1, 1, 1><<<grid, nw * 32, 0, stream>>>(a);
    else
      conv2d_wmma_kernel<3, 3, 1><<<grid, nw * 32, 0, stream>>>(a);
  };

  // ResNet basic block (all convs stride 1 at 92x160)
  auto run_block = [&](int base, bool down, int ci, int cm, int co,
                       const float* xin, float* htmp, float* sctmp, float* out) {
    const __bf16* w1p = pack(F(base + 0), cm, ci * 9);
    const __bf16* w2p = pack(F(base + 4), co, cm * 9);
    const float* resp = xin;
    if (down) {
      const __bf16* wdp = pack(F(base + 8), co, ci);
      conv(xin, wdp, F(base + 9), F(base + 10), F(base + 11), nullptr, sctmp,
           ci, 92, 160, co, 92, 160, 1, 1, 0, CF_BIAS | CF_BN);
      resp = sctmp;
    }
    conv(xin, w1p, F(base + 1), F(base + 2), F(base + 3), nullptr, htmp,
         ci, 92, 160, cm, 92, 160, 3, 1, 1, CF_BIAS | CF_BN | CF_RELU);
    conv(htmp, w2p, F(base + 5), F(base + 6), F(base + 7), resp, out,
         cm, 92, 160, co, 92, 160, 3, 1, 1, CF_BIAS | CF_BN | CF_RES | CF_RELU);
  };

  // ---- stem: 3->64, 7x7 s2 p3 + BN + ReLU : [2,3,368,640] -> P0 [2,64,184,320]
  conv(F(0), pack(F(1), 64, 3 * 49), F(2), F(3), F(4), nullptr, P0,
       3, 368, 640, 64, 184, 320, 7, 2, 3, CF_BIAS | CF_BN | CF_RELU);

  // ---- maxpool 3x3 s2 p1 -> P1 [2,64,92,160]
  {
    int total = 2 * 64 * 92 * 160;
    maxpool_kernel<<<(total + 255) / 256, 256, 0, stream>>>(P0, P1, 2 * 64, 184, 320,
                                                            92, 160);
  }

  // ---- residual stages (all at 92x160)
  run_block( 5, true,   64,  64,  64, P1, P3, P2, P0);
  run_block(17, false,  64,  64,  64, P0, P3, nullptr, P1);
  run_block(25, true,   64,  64, 128, P1, P3, P2, P0);
  run_block(37, false, 128,  64, 128, P0, P3, nullptr, P1);
  run_block(45, true,  128, 128, 256, P1, P3, P2, P0);
  run_block(57, false, 256, 128, 256, P0, P3, nullptr, P1);
  run_block(65, true,  256, 256, 512, P1, P3, P2, P0);
  run_block(77, false, 512, 256, 512, P0, P3, nullptr, P1);

  // ---- dim_set: 512->128 + ReLU -> P0
  conv(P1, pack(F(85), 128, 512 * 9), F(86), nullptr, nullptr, nullptr, P0,
       512, 92, 160, 128, 92, 160, 3, 1, 1, CF_BIAS | CF_RELU);

  // ---- SCNN sequential passes (in place on P0)
  const __bf16* wpd = pack(F(87), 128, 1152);
  const __bf16* wpu = pack(F(88), 128, 1152);
  const __bf16* wpr = pack(F(89), 128, 1152);
  const __bf16* wpl = pack(F(90), 128, 1152);
  for (int i = 1; i <= 91; ++i)
    scnn_step_kernel<<<20, 256, 0, stream>>>(P0, wpd, 1, i - 1, i);
  for (int i = 90; i >= 1; --i)
    scnn_step_kernel<<<20, 256, 0, stream>>>(P0, wpu, 1, i + 1, i);
  for (int j = 1; j <= 159; ++j)
    scnn_step_kernel<<<12, 256, 0, stream>>>(P0, wpr, 0, j - 1, j);
  for (int j = 158; j >= 1; --j)
    scnn_step_kernel<<<12, 256, 0, stream>>>(P0, wpl, 0, j + 1, j);

  // ---- dim_set_re: 128->512 + ReLU -> P1
  conv(P0, pack(F(91), 512, 128 * 9), F(92), nullptr, nullptr, nullptr, P1,
       128, 92, 160, 512, 92, 160, 3, 1, 1, CF_BIAS | CF_RELU);

  // ---- dec1: 512->256 + ReLU -> P2
  conv(P1, pack(F(93), 256, 512 * 9), F(94), nullptr, nullptr, nullptr, P2,
       512, 92, 160, 256, 92, 160, 3, 1, 1, CF_BIAS | CF_RELU);

  // ---- dec3: convT 256->128 (x2 upsample) + ReLU -> P0 [2,128,184,320]
  {
    int total = 2 * 128 * 184 * 320;
    convt_kernel<<<(total + 255) / 256, 256, 0, stream>>>(P2, F(95), F(96), P0,
                                                          2, 256, 92, 160, 128, 184,
                                                          320, 1);
  }
  // conv 128->64 + ReLU -> P1
  conv(P0, pack(F(97), 64, 128 * 9), F(98), nullptr, nullptr, nullptr, P1,
       128, 184, 320, 64, 184, 320, 3, 1, 1, CF_BIAS | CF_RELU);
  // conv 64->32 + ReLU -> P2
  conv(P1, pack(F(99), 32, 64 * 9), F(100), nullptr, nullptr, nullptr, P2,
       64, 184, 320, 32, 184, 320, 3, 1, 1, CF_BIAS | CF_RELU);
  // conv 32->32 + ReLU -> P3
  conv(P2, pack(F(101), 32, 32 * 9), F(102), nullptr, nullptr, nullptr, P3,
       32, 184, 320, 32, 184, 320, 3, 1, 1, CF_BIAS | CF_RELU);
  // convT 32->2 (x2 upsample) + ReLU -> d_out [2,2,368,640]
  {
    int total = 2 * 2 * 368 * 640;
    convt_kernel<<<(total + 255) / 256, 256, 0, stream>>>(P3, F(103), F(104),
                                                          (float*)d_out,
                                                          2, 32, 184, 320, 2, 368,
                                                          640, 1);
  }
}
